// ImpEntEncoderFast_46102178955798
// MI455X (gfx1250) — matrix operand order repaired and checked by reading, not compiled, not run on hardware
//
#include <hip/hip_runtime.h>
#include <hip/hip_bf16.h>

// ---------------------------------------------------------------------------
// PixelCNN masked-conv entropy encoder for MI455X (gfx1250, wave32, WMMA).
// Strategy: implicit-GEMM masked conv via v_wmma_f32_16x16x32_bf16.
//   - activations: bf16, zero-halo padded layout [(y+2)*Wp + (x+2)]*160 + c
//   - weights: device-side repack into WMMA B-matrix lane layout, LDS-staged
//   - per wave: 16 pixels x 144 out-channels accumulated in 9 v8f tiles
// ---------------------------------------------------------------------------

typedef __attribute__((ext_vector_type(16))) __bf16 v16bf;
typedef __attribute__((ext_vector_type(8)))  float  v8f;
typedef __attribute__((ext_vector_type(4)))  unsigned int u32x4;

union V16 { v16bf v; u32x4 q[2]; };

constexpr int Hh = 256, Ww = 512, HW = Hh * Ww;
constexpr int CPG = 144;          // hidden channels
constexpr int CP  = 160;          // padded to 5 x 32 (WMMA K chunks)
constexpr int NOUTC = 49;
constexpr int Hp = Hh + 2;        // 2-row zero halo on top
constexpr int Wp = Ww + 4;        // 2-col zero halo left+right
constexpr float SCALEV = 2.0f / 47.0f;
constexpr float TABLE_TOTAL = 65536.0f;

// causal 5x5 mask-B taps in raster order (first 12 == mask-A taps)
__constant__ int c_kh[13] = {0,0,0,0,0, 1,1,1,1,1, 2,2,2};
__constant__ int c_kw[13] = {0,1,2,3,4, 0,1,2,3,4, 0,1,2};

// workspace layout (bytes)
constexpr size_t ALN(size_t x) { return (x + 255) & ~size_t(255); }
constexpr size_t SZ_ACT_E = (size_t)Hp * Wp * CP;        // bf16 elements
constexpr size_t SZ_ACT_B = SZ_ACT_E * 2;
constexpr size_t LAYER_E  = (size_t)13 * 5 * 9 * 512;    // packed bf16/layer
constexpr size_t FIN_E    = (size_t)13 * 5 * 4 * 512;
constexpr size_t OFF_X  = 0;
constexpr size_t OFF_H  = ALN(OFF_X + SZ_ACT_B);
constexpr size_t OFF_WH = ALN(OFF_H + SZ_ACT_B);
constexpr size_t OFF_WF = ALN(OFF_WH + 10 * LAYER_E * 2);
constexpr size_t OFF_LG = ALN(OFF_WF + FIN_E * 2);

// ---------------------------------------------------------------------------
__global__ void zero_bf16(__bf16* p, size_t n16) {  // n16 = count of 16B chunks
  size_t i = (size_t)blockIdx.x * blockDim.x + threadIdx.x;
  u32x4* d = (u32x4*)p;
  u32x4 z = {0u, 0u, 0u, 0u};
  for (; i < n16; i += (size_t)gridDim.x * blockDim.x) d[i] = z;
}

// repack hidden weights -> B-matrix lane layout:
// wp[((((l*13+t)*5+q)*9+nb)<<9) + lane*16 + e], k = (lane>>4)*16 + e
__global__ void repack_hidden(const float* __restrict__ wres, __bf16* __restrict__ wp) {
  int idx = blockIdx.x * blockDim.x + threadIdx.x;
  if (idx >= (int)(10 * LAYER_E)) return;
  int e    = idx & 511;
  int rest = idx >> 9;
  int nb = rest % 9;  rest /= 9;
  int q  = rest % 5;  rest /= 5;
  int t  = rest % 13;
  int l  = rest / 13;
  int lane = e >> 4, wi = e & 15;
  int n  = lane & 15;
  int k  = ((lane >> 4) << 4) + wi;
  int ic = q * 32 + k;
  int oc = nb * 16 + n;
  float v = 0.0f;
  if (ic < CPG)
    v = wres[(((size_t)l * CPG + oc) * CPG + ic) * 25 + c_kh[t] * 5 + c_kw[t]];
  wp[idx] = (__bf16)v;
}

__global__ void repack_final(const float* __restrict__ wl, __bf16* __restrict__ wp) {
  int idx = blockIdx.x * blockDim.x + threadIdx.x;
  if (idx >= (int)FIN_E) return;
  int e    = idx & 511;
  int rest = idx >> 9;
  int nb = rest % 4;  rest /= 4;
  int q  = rest % 5;  rest /= 5;
  int t  = rest;
  int lane = e >> 4, wi = e & 15;
  int n  = lane & 15;
  int k  = ((lane >> 4) << 4) + wi;
  int ic = q * 32 + k;
  int oc = nb * 16 + n;
  float v = 0.0f;
  if (ic < CPG && oc < NOUTC)
    v = wl[((size_t)oc * CPG + ic) * 25 + c_kh[t] * 5 + c_kw[t]];
  wp[idx] = (__bf16)v;
}

// layer 0: 1 -> 144 masked conv (mask A = first 12 taps), scale + relu
__global__ void layer0(const float* __restrict__ data, const float* __restrict__ w0,
                       const float* __restrict__ b0, __bf16* __restrict__ X) {
  int idx = blockIdx.x * blockDim.x + threadIdx.x;
  if (idx >= HW * CPG) return;
  int c = idx % CPG;
  int p = idx / CPG;
  int y = p >> 9, x = p & 511;
  float acc = b0[c];
#pragma unroll
  for (int t = 0; t < 12; ++t) {
    int yy = y + c_kh[t] - 2;
    int xx = x + c_kw[t] - 2;
    if (yy >= 0 && xx >= 0 && xx < Ww) {
      float v = data[yy * Ww + xx] * SCALEV - 1.0f;
      acc += v * w0[c * 25 + c_kh[t] * 5 + c_kw[t]];
    }
  }
  acc = fmaxf(acc, 0.0f);
  X[((size_t)(y + 2) * Wp + (x + 2)) * CP + c] = (__bf16)acc;
}

// ---------------------------------------------------------------------------
// Masked 5x5 conv 144->16*NB as implicit GEMM over 13 taps x 5 K-chunks.
// One wave = 16-pixel row tile x 16*NB out channels (NB v8f accumulators).
// Weights staged per-tap in LDS, shared by the block's 8 waves.
// ---------------------------------------------------------------------------
template <int NB, bool RELU, bool RESID, bool F32OUT, int COUT>
__global__ __launch_bounds__(256) void
mconv_wmma(const __bf16* __restrict__ actIn, const __bf16* __restrict__ wpack,
           const float* __restrict__ bias, __bf16* __restrict__ actOut,
           float* __restrict__ logits) {
  __shared__ __attribute__((aligned(16))) __bf16 sw[5 * NB * 512];

  const int lane = threadIdx.x & 31;
  const int ln15 = lane & 15;           // A-row (pixel), B/C column (out ch)
  const int half = lane >> 4;           // K-half selector
  const int wave = blockIdx.x * (blockDim.x >> 5) + (threadIdx.x >> 5);
  const int y  = wave >> 5;             // 32 tiles per image row
  const int x0 = (wave & 31) << 4;

  v8f acc[NB];
#pragma unroll
  for (int nb = 0; nb < NB; ++nb) {
    int c = nb * 16 + ln15;
    float bv = (c < COUT) ? bias[c] : 0.0f;
#pragma unroll
    for (int r = 0; r < 8; ++r) acc[nb][r] = bv;
  }

  const int nvec = (5 * NB * 512) / 8;  // u32x4 chunks of one tap's weights
  for (int t = 0; t < 13; ++t) {
    __syncthreads();                    // waves done reading previous tap
    const u32x4* src = (const u32x4*)(wpack + (size_t)t * 5 * NB * 512);
    u32x4* dst = (u32x4*)sw;
    for (int i = threadIdx.x; i < nvec; i += blockDim.x) dst[i] = src[i];
    if (t + 1 < 13)                     // warm L2/L1 for the next tap
      __builtin_prefetch(wpack + (size_t)(t + 1) * 5 * NB * 512, 0, 1);
    __syncthreads();

    const int dy = c_kh[t] - 2, dx = c_kw[t] - 2;
    const __bf16* abase =
        actIn + ((size_t)(y + dy + 2) * Wp + (x0 + ln15 + dx + 2)) * CP + half * 16;
#pragma unroll
    for (int q = 0; q < 5; ++q) {
      V16 A;
      A.q[0] = *(const u32x4*)(abase + q * 32);        // K = half*16 .. +7
      A.q[1] = *(const u32x4*)(abase + q * 32 + 8);    // K = half*16+8 .. +7
#pragma unroll
      for (int nb = 0; nb < NB; ++nb) {
        V16 B;
        const __bf16* bp = sw + ((q * NB + nb) * 512 + lane * 16);
        B.q[0] = *(const u32x4*)(bp);
        B.q[1] = *(const u32x4*)(bp + 8);
        acc[nb] = __builtin_amdgcn_wmma_f32_16x16x32_bf16(
            false, A.v, false, B.v, (short)0, acc[nb], false, false);
      }
    }
  }

#pragma unroll
  for (int nb = 0; nb < NB; ++nb) {
    int c = nb * 16 + ln15;
#pragma unroll
    for (int r = 0; r < 8; ++r) {
      float v = acc[nb][r];
      if (RELU) v = fmaxf(v, 0.0f);
      int mm = r + half * 8;            // pixel within the 16-wide tile
      if (F32OUT) {
        logits[((size_t)(y * Ww + x0 + mm)) * 64 + c] = v;
      } else {
        size_t o = ((size_t)(y + 2) * Wp + (x0 + mm + 2)) * CP + c;
        if (RESID) v += (float)actOut[o];
        actOut[o] = (__bf16)v;
      }
    }
  }
}

// per-pixel softmax -> cumulative frequency table (channel-major output)
__global__ void softmax_cdf(const float* __restrict__ logits, float* __restrict__ out) {
  int p = blockIdx.x * blockDim.x + threadIdx.x;
  if (p >= HW) return;
  const float* l = logits + (size_t)p * 64;
  float v[NOUTC];
  float mx = -1e30f;
#pragma unroll
  for (int i = 0; i < NOUTC; ++i) { v[i] = l[i]; mx = fmaxf(mx, v[i]); }
  float s = 0.0f;
#pragma unroll
  for (int i = 0; i < NOUTC; ++i) { v[i] = __expf(v[i] - mx); s += v[i]; }
  float inv = TABLE_TOTAL / s;
  out[p] = 0.0f;
  float cum = 0.0f;
#pragma unroll
  for (int i = 0; i < NOUTC; ++i) {
    cum += v[i];
    out[(size_t)(i + 1) * HW + p] = cum * inv;
  }
}

// ---------------------------------------------------------------------------
extern "C" void kernel_launch(void* const* d_in, const int* in_sizes, int n_in,
                              void* d_out, int out_size, void* d_ws, size_t ws_size,
                              hipStream_t stream) {
  (void)in_sizes; (void)n_in; (void)out_size; (void)ws_size;
  const float* data = (const float*)d_in[0];
  const float* w0   = (const float*)d_in[1];
  const float* b0   = (const float*)d_in[2];
  const float* wres = (const float*)d_in[3];
  const float* bres = (const float*)d_in[4];
  const float* wl   = (const float*)d_in[5];
  const float* bl   = (const float*)d_in[6];

  char* ws = (char*)d_ws;
  __bf16* X  = (__bf16*)(ws + OFF_X);
  __bf16* H1 = (__bf16*)(ws + OFF_H);
  __bf16* WH = (__bf16*)(ws + OFF_WH);
  __bf16* WF = (__bf16*)(ws + OFF_WF);
  float*  LG = (float*)(ws + OFF_LG);

  // zero activation buffers (halo + channel padding must be 0 every call)
  zero_bf16<<<2048, 256, 0, stream>>>(X,  SZ_ACT_B / 16);
  zero_bf16<<<2048, 256, 0, stream>>>(H1, SZ_ACT_B / 16);

  // repack weights fp32 -> bf16 WMMA B-matrix layout
  repack_hidden<<<(int)((10 * LAYER_E + 255) / 256), 256, 0, stream>>>(wres, WH);
  repack_final<<<(int)((FIN_E + 255) / 256), 256, 0, stream>>>(wl, WF);

  // layer 0 (mask A, 1 input channel): scalar path, bandwidth-trivial
  layer0<<<(HW * CPG + 255) / 256, 256, 0, stream>>>(data, w0, b0, X);

  // 5 residual blocks: h1 = relu(conv(x)); x = x + relu(conv(h1))
  for (int blk = 0; blk < 5; ++blk) {
    const __bf16* wA = WH + (size_t)(2 * blk) * LAYER_E;
    const __bf16* wB = WH + (size_t)(2 * blk + 1) * LAYER_E;
    const float* bA = bres + (2 * blk) * CPG;
    const float* bB = bres + (2 * blk + 1) * CPG;
    mconv_wmma<9, true, false, false, CPG><<<1024, 256, 0, stream>>>(X, wA, bA, H1, nullptr);
    mconv_wmma<9, true, true,  false, CPG><<<1024, 256, 0, stream>>>(H1, wB, bB, X, nullptr);
  }

  // final 144 -> 49 logits conv (padded to 64 channels), fp32 out
  mconv_wmma<4, false, false, true, NOUTC><<<1024, 256, 0, stream>>>(X, WF, bl, nullptr, LG);

  // softmax -> CDF table, channel-major (1, 50, H, W)
  softmax_cdf<<<HW / 256, 256, 0, stream>>>(LG, (float*)d_out);
}
